// PoolingONCDEClassifier_8426725834811
// MI455X (gfx1250) — matrix operand order repaired and checked by reading, not compiled, not run on hardware
//
#include <hip/hip_runtime.h>

// ---------------------------------------------------------------------------
// PoolingONCDEClassifier for MI455X (gfx1250, wave32, WMMA f16->f32)
//
//   1) prep_weights : transpose + fp32->f16 convert weights into d_ws
//   2) cde_integrate: 8 blocks x 1024 thr (one block per 16-row batch tile),
//                     511 intervals x 4 Heun substeps x 2 f-evals.
//                     f16 state mirrors in LDS -> A-frags are pure ds_load_b128;
//                     GEMM2 A-frags hoisted into registers across N-tiles;
//                     branchless softplus/tanh (no exec-mask churn in hot loop).
//   3) pool         : batch-mean of reps -> ctx[:,:,128:256]
//   4) classifier   : [65536,256]x[256,256] leaky_relu then x[256,16] WMMA GEMM
// ---------------------------------------------------------------------------

#define B_   128
#define L_   512
#define LI_  512
#define D_   16
#define H_   128
#define W_   128
#define WC_  256
#define C_   10
#define KSUB 4
#define EPS_ 1e-2f
#define NT_  1024   // threads per integrate block (32 waves)

typedef __attribute__((ext_vector_type(16))) _Float16 v16h;
typedef __attribute__((ext_vector_type(8)))  float    v8f;

// workspace layout (halves)
#define WS_VW0T 0                         // [128][128]
#define WS_VW1T (WS_VW0T + H_*W_)         // [128][2048]
#define WS_CW0T (WS_VW1T + W_*H_*D_)      // [256][256]
#define WS_CW1T (WS_CW0T + 2*H_*WC_)      // [256][16]
#define WS_TOTAL (WS_CW1T + WC_*16)

// gfx1250 has V_TANH_F32 (TRANS). Use the builtin when declared, else a
// branchless exp-based fallback (saturates via inf -> 2/inf = 0).
#if __has_builtin(__builtin_amdgcn_tanhf)
__device__ __forceinline__ float fast_tanhf(float x) { return __builtin_amdgcn_tanhf(x); }
#elif __has_builtin(__builtin_amdgcn_tanh_f32)
__device__ __forceinline__ float fast_tanhf(float x) { return __builtin_amdgcn_tanh_f32(x); }
#else
__device__ __forceinline__ float fast_tanhf(float x) {
  return 1.f - 2.f * __frcp_rn(__expf(2.f * x) + 1.f);
}
#endif

// branchless, numerically stable softplus: max(x,0) + log1p(exp(-|x|))
__device__ __forceinline__ float softplusf(float x) {
  return fmaxf(x, 0.f) + __logf(1.f + __expf(-fabsf(x)));
}

// A-fragment (16x32 f16, MxK) per ISA layout, gathered from an f32 matrix.
__device__ __forceinline__ v16h load_afrag_f32(const float* src, int ld, int kc, int lane) {
  const int m    = lane & 15;
  const int base = ((lane >> 4) << 3) + (kc << 5);
  const float* r = src + m * ld + base;
  v16h a;
#pragma unroll
  for (int e = 0; e < 8; ++e) a[e] = (_Float16)r[e];
#pragma unroll
  for (int e = 0; e < 8; ++e) a[8 + e] = (_Float16)r[16 + e];
  return a;
}

// Same, from an f16 matrix: two contiguous 16B ds/global loads, no converts.
__device__ __forceinline__ v16h load_afrag_f16(const _Float16* src, int ld, int kc, int lane) {
  const int m    = lane & 15;
  const int base = ((lane >> 4) << 3) + (kc << 5);
  const _Float16* r = src + m * ld + base;
  v16h a;
#pragma unroll
  for (int e = 0; e < 8; ++e) a[e] = r[e];
#pragma unroll
  for (int e = 0; e < 8; ++e) a[8 + e] = r[16 + e];
  return a;
}

// B-fragment (32x16 f16, KxN): lane holds column N = n0+lane%16, K in same order as A-frag.
__device__ __forceinline__ v16h load_bfrag(const _Float16* Bt, int ld, int kc, int n0, int lane) {
  const int n  = n0 + (lane & 15);
  const int kb = (kc << 5) + ((lane >> 4) << 3);
  v16h b;
#pragma unroll
  for (int e = 0; e < 8; ++e) b[e] = Bt[(size_t)(kb + e) * ld + n];
#pragma unroll
  for (int e = 0; e < 8; ++e) b[8 + e] = Bt[(size_t)(kb + 16 + e) * ld + n];
  return b;
}

__device__ __forceinline__ v8f wmma_f16(v16h a, v16h b, v8f c) {
  return __builtin_amdgcn_wmma_f32_16x16x32_f16(false, a, false, b, (short)0, c, false, false);
}

// ---------------------------------------------------------------------------
// 1) weight prep: fp32 -> f16 transposed images in workspace
// ---------------------------------------------------------------------------
__global__ void prep_weights(const float* __restrict__ vw0, const float* __restrict__ vw1,
                             const float* __restrict__ cw0, const float* __restrict__ cw1,
                             _Float16* __restrict__ ws) {
  int i = blockIdx.x * blockDim.x + threadIdx.x;
  const int N0 = H_ * W_;         // 16384  vw0t  [K=H][N=W]
  const int N1 = W_ * H_ * D_;    // 262144 vw1t  [K=W][N=H*D]
  const int N2 = 2 * H_ * WC_;    // 65536  cw0t  [K=2H][N=WC]
  const int N3 = WC_ * 16;        // 4096   cw1t  [K=WC][N=16] (C=10 padded)
  if (i < N0) {
    int k = i >> 7, n = i & 127;                      // vw0 is (W,H)
    ws[WS_VW0T + i] = (_Float16)vw0[n * H_ + k];
  } else if (i < N0 + N1) {
    int j = i - N0; int k = j >> 11, n = j & 2047;    // vw1 is (H*D, W)
    ws[WS_VW1T + j] = (_Float16)vw1[(size_t)n * W_ + k];
  } else if (i < N0 + N1 + N2) {
    int j = i - N0 - N1; int k = j >> 8, n = j & 255; // cw0 is (WC, 2H)
    ws[WS_CW0T + j] = (_Float16)cw0[n * (2 * H_) + k];
  } else if (i < N0 + N1 + N2 + N3) {
    int j = i - N0 - N1 - N2; int k = j >> 4, n = j & 15; // cw1 is (C, WC)
    ws[WS_CW1T + j] = (_Float16)((n < C_) ? cw1[n * WC_ + k] : 0.f);
  }
}

// ---------------------------------------------------------------------------
// 2) CDE integration: one block per 16 batch rows, 1024 threads = 32 waves
// ---------------------------------------------------------------------------
__device__ void cde_f(float tq,
                      const _Float16* __restrict__ yin_h, // LDS [16][128] f16 mirror
                      float* __restrict__ dyout,          // LDS [16][128] f32
                      const float* __restrict__ ts_interp,
                      const float* __restrict__ obs_interp,
                      const _Float16* __restrict__ vw0t,
                      const _Float16* __restrict__ vw1t,
                      const float* __restrict__ s_vb0,
                      const float* __restrict__ s_vb1,
                      _Float16* __restrict__ s_ah,        // LDS [16][128] f16
                      float* __restrict__ s_dxdt,         // LDS [16][16]
                      int* __restrict__ s_idx,            // LDS [16]
                      int b0, int tid, int lane, int wv) {
  // piecewise-constant dX/dt at tq (monotone cursor per row)
  if (tid < 16) {
    const int b = b0 + tid;
    const float* r = ts_interp + (size_t)b * LI_;
    int idx = s_idx[tid];
    while (idx < LI_ - 2 && r[idx + 1] <= tq) ++idx;
    s_idx[tid] = idx;
    const float inv = 1.f / fmaxf(r[idx + 1] - r[idx], EPS_);
    const float* p0 = obs_interp + ((size_t)b * LI_ + idx) * D_;
#pragma unroll
    for (int d = 0; d < D_; ++d) s_dxdt[tid * 16 + d] = (p0[D_ + d] - p0[d]) * inv;
  }
  __syncthreads();

  // GEMM1: a = softplus(y @ vw0^T + vb0)   [16 x 128], waves 0..7, 1 N-tile each
  if (wv < 8) {
    const int j = wv;
    v8f acc = {};
#pragma unroll
    for (int kc = 0; kc < 4; ++kc) {
      v16h af = load_afrag_f16(yin_h, H_, kc, lane);
      v16h bf = load_bfrag(vw0t, W_, kc, j * 16, lane);
      acc = wmma_f16(af, bf, acc);
    }
    const int n    = j * 16 + (lane & 15);
    const int mofs = (lane >> 4) << 3;
    const float bias = s_vb0[n];
#pragma unroll
    for (int r = 0; r < 8; ++r)
      s_ah[(r + mofs) * W_ + n] = (_Float16)softplusf(acc[r] + bias);
  }
  __syncthreads();

  // GEMM2 + D=16 contraction. A-fragments identical for all N-tiles: hoist.
  v16h af2[4];
#pragma unroll
  for (int kc = 0; kc < 4; ++kc) af2[kc] = load_afrag_f16(s_ah, W_, kc, lane);

  const int dcol = lane & 15;
  const int mofs = (lane >> 4) << 3;
#pragma unroll 1
  for (int jj = 0; jj < 4; ++jj) {
    const int j = wv + jj * 32;   // N-tile 0..127  (== h index)
    v8f acc = {};
#pragma unroll
    for (int kc = 0; kc < 4; ++kc)
      acc = wmma_f16(af2[kc], load_bfrag(vw1t, H_ * D_, kc, j * 16, lane), acc);
    const float bias = s_vb1[j * 16 + dcol];
#pragma unroll
    for (int r = 0; r < 8; ++r) {
      const int m = r + mofs;
      float prod = fast_tanhf(acc[r] + bias) * s_dxdt[m * 16 + dcol];
      prod += __shfl_xor(prod, 1, 32);
      prod += __shfl_xor(prod, 2, 32);
      prod += __shfl_xor(prod, 4, 32);
      prod += __shfl_xor(prod, 8, 32);
      if (dcol == 0) dyout[m * H_ + j] = prod;
    }
  }
  __syncthreads();
}

__global__ __launch_bounds__(NT_)
void cde_integrate(const float* __restrict__ ts,
                   const float* __restrict__ ts_interp,
                   const float* __restrict__ obs_interp,
                   const float* __restrict__ iw0, const float* __restrict__ ib0,
                   const float* __restrict__ iw1, const float* __restrict__ ib1,
                   const float* __restrict__ vb0, const float* __restrict__ vb1,
                   const _Float16* __restrict__ vw0t,
                   const _Float16* __restrict__ vw1t,
                   const int* __restrict__ t_max_p,
                   float* __restrict__ ctx_out /* [B][L][2H], first H filled here */) {
  __shared__ float s_vb0[W_];
  __shared__ float s_vb1[H_ * D_];
  __shared__ __align__(16) float s_y[16 * H_];
  __shared__ __align__(16) float s_k1[16 * H_];
  __shared__ __align__(16) float s_yt[16 * H_];
  __shared__ __align__(16) float s_dy[16 * H_];
  __shared__ __align__(32) _Float16 s_yh[16 * H_];   // f16 mirror of y
  __shared__ __align__(32) _Float16 s_yth[16 * H_];  // f16 mirror of ytmp
  __shared__ __align__(32) _Float16 s_ah[16 * W_];
  __shared__ float s_dxdt[16 * 16];
  __shared__ float s_x0[16 * 16];
  __shared__ int   s_idx[16];

  const int tid  = threadIdx.x;
  const int lane = tid & 31;
  const int wv   = tid >> 5;
  const int b0   = blockIdx.x * 16;
  const float tmax = (float)(*t_max_p);

  for (int i = tid; i < W_; i += NT_) s_vb0[i] = vb0[i];
  for (int i = tid; i < H_ * D_; i += NT_) s_vb1[i] = vb1[i];
  __syncthreads();

  // ---- initial state: x0 = X(ts[0]); y0 = iw1 @ softplus(iw0 @ x0 + ib0) + ib1
  if (tid < 16) {
    const int b = b0 + tid;
    const float* r = ts_interp + (size_t)b * LI_;
    const float t0q = ts[0];
    int idx = 0;
    while (idx < LI_ - 2 && r[idx + 1] <= t0q) ++idx;
    const float t0 = r[idx];
    const float dt = fmaxf(r[idx + 1] - t0, EPS_);
    const float w  = (t0q - t0) / dt;
    const float* p0 = obs_interp + ((size_t)b * LI_ + idx) * D_;
#pragma unroll
    for (int d = 0; d < D_; ++d) s_x0[tid * 16 + d] = p0[d] + w * (p0[D_ + d] - p0[d]);
    s_idx[tid] = idx;
  }
  __syncthreads();
  for (int o = tid; o < 16 * W_; o += NT_) {         // hidden -> s_yt (scratch)
    const int m = o >> 7, wc = o & 127;
    float acc = ib0[wc];
    const float* x = &s_x0[m * 16];
    const float* wr = iw0 + wc * D_;
#pragma unroll
    for (int d = 0; d < D_; ++d) acc += x[d] * wr[d];
    s_yt[m * W_ + wc] = softplusf(acc);
  }
  __syncthreads();
  for (int o = tid; o < 16 * H_; o += NT_) {         // y0 (+ f16 mirror)
    const int m = o >> 7, hh = o & 127;
    float acc = ib1[hh];
    const float* hv = &s_yt[m * W_];
    const float* wr = iw1 + hh * W_;
    for (int wc = 0; wc < W_; ++wc) acc += hv[wc] * wr[wc];
    s_y[o]  = acc;
    s_yh[o] = (_Float16)acc;
  }
  __syncthreads();

  // save reps[l=0] (masked) into ctx[:, 0, 0:H]
  {
    const bool keep = (ts[0] <= tmax);
    for (int o = tid; o < 16 * H_; o += NT_) {
      const int m = o >> 7, hh = o & 127;
      ctx_out[((size_t)(b0 + m) * L_ + 0) * (2 * H_) + hh] = keep ? s_y[o] : -99.f;
    }
  }
  __syncthreads();

  // ---- time loop: 511 intervals x 4 Heun substeps
  for (int i = 0; i < L_ - 1; ++i) {
    const float ta = ts[i];
    const float h  = (ts[i + 1] - ta) * (1.f / KSUB);
    for (int k = 0; k < KSUB; ++k) {
      const float t = ta + k * h;
      cde_f(t, s_yh, s_k1, ts_interp, obs_interp, vw0t, vw1t,
            s_vb0, s_vb1, s_ah, s_dxdt, s_idx, b0, tid, lane, wv);
      for (int o = tid; o < 16 * H_; o += NT_) {
        const float v = s_y[o] + h * s_k1[o];
        s_yt[o]  = v;
        s_yth[o] = (_Float16)v;
      }
      __syncthreads();
      cde_f(t + h, s_yth, s_dy, ts_interp, obs_interp, vw0t, vw1t,
            s_vb0, s_vb1, s_ah, s_dxdt, s_idx, b0, tid, lane, wv);
      for (int o = tid; o < 16 * H_; o += NT_) {
        const float v = s_y[o] + 0.5f * h * (s_k1[o] + s_dy[o]);
        s_y[o]  = v;
        s_yh[o] = (_Float16)v;
      }
      __syncthreads();
    }
    const bool keep = (ts[i + 1] <= tmax);
    for (int o = tid; o < 16 * H_; o += NT_) {
      const int m = o >> 7, hh = o & 127;
      ctx_out[((size_t)(b0 + m) * L_ + (i + 1)) * (2 * H_) + hh] = keep ? s_y[o] : -99.f;
    }
    __syncthreads();
  }
}

// ---------------------------------------------------------------------------
// 3) batch pool: ctx[:,:,H:2H] = sum_b reps / sum_b valid_mask
// ---------------------------------------------------------------------------
__global__ __launch_bounds__(128)
void pool_kernel(const float* __restrict__ valid_mask, float* __restrict__ ctx) {
  const int l  = blockIdx.x;
  const int hh = threadIdx.x;
  float s = 0.f, denom = 0.f;
  for (int b = 0; b < B_; ++b) s += ctx[((size_t)b * L_ + l) * (2 * H_) + hh];
  for (int b = 0; b < B_; ++b) denom += valid_mask[(size_t)b * L_ + l];
  const float p = s / denom;
  for (int b = 0; b < B_; ++b) ctx[((size_t)b * L_ + l) * (2 * H_) + H_ + hh] = p;
}

// ---------------------------------------------------------------------------
// 4) classifier: logits = leaky_relu(ctx @ cw0^T + cb0) @ cw1^T + cb1
//    128 threads = 4 waves, one 16-row M-tile per wave, 1024 blocks
// ---------------------------------------------------------------------------
__global__ __launch_bounds__(128)
void classifier_kernel(const float* __restrict__ ctx,
                       const _Float16* __restrict__ cw0t,
                       const _Float16* __restrict__ cw1t,
                       const float* __restrict__ cb0,
                       const float* __restrict__ cb1,
                       float* __restrict__ logits) {
  __shared__ __align__(32) _Float16 s_hid[4][16 * WC_];   // 32 KB

  const int tid  = threadIdx.x;
  const int lane = tid & 31;
  const int wv   = tid >> 5;
  const int tile = blockIdx.x * 4 + wv;                   // 0..4095
  const float* arow = ctx + (size_t)tile * 16 * (2 * H_);

  // keep all 8 A K-chunk fragments in registers, reuse across 16 N-tiles
  v16h af[8];
#pragma unroll
  for (int kc = 0; kc < 8; ++kc) af[kc] = load_afrag_f32(arow, 2 * H_, kc, lane);

  _Float16* hid = &s_hid[wv][0];
  const int ncol = lane & 15;
  const int mofs = (lane >> 4) << 3;
#pragma unroll 1
  for (int j = 0; j < 16; ++j) {
    v8f acc = {};
#pragma unroll
    for (int kc = 0; kc < 8; ++kc)
      acc = wmma_f16(af[kc], load_bfrag(cw0t, WC_, kc, j * 16, lane), acc);
    const int n = j * 16 + ncol;
    const float bias = cb0[n];
#pragma unroll
    for (int r = 0; r < 8; ++r) {
      const float x = acc[r] + bias;
      hid[(r + mofs) * WC_ + n] = (_Float16)((x > 0.f) ? x : 0.01f * x);  // leaky_relu
    }
  }
  __syncthreads();

  // second layer: [16,256] @ [256,16(pad)] -> one output tile
  v8f acc2 = {};
#pragma unroll
  for (int kc = 0; kc < 8; ++kc) {
    v16h a2 = load_afrag_f16(hid, WC_, kc, lane);
    v16h bf = load_bfrag(cw1t, 16, kc, 0, lane);
    acc2 = wmma_f16(a2, bf, acc2);
  }
  if (ncol < C_) {
    const float bias = cb1[ncol];
#pragma unroll
    for (int r = 0; r < 8; ++r) {
      const int row = tile * 16 + r + mofs;
      logits[(size_t)row * C_ + ncol] = acc2[r] + bias;
    }
  }
}

// ---------------------------------------------------------------------------
extern "C" void kernel_launch(void* const* d_in, const int* in_sizes, int n_in,
                              void* d_out, int out_size, void* d_ws, size_t ws_size,
                              hipStream_t stream) {
  const float* ts         = (const float*)d_in[0];
  const float* ts_interp  = (const float*)d_in[1];
  const float* obs_interp = (const float*)d_in[2];
  const float* valid_mask = (const float*)d_in[3];
  const float* iw0 = (const float*)d_in[4];
  const float* ib0 = (const float*)d_in[5];
  const float* iw1 = (const float*)d_in[6];
  const float* ib1 = (const float*)d_in[7];
  const float* vw0 = (const float*)d_in[8];
  const float* vb0 = (const float*)d_in[9];
  const float* vw1 = (const float*)d_in[10];
  const float* vb1 = (const float*)d_in[11];
  const float* cw0 = (const float*)d_in[12];
  const float* cb0 = (const float*)d_in[13];
  const float* cw1 = (const float*)d_in[14];
  const float* cb1 = (const float*)d_in[15];
  const int*   t_max = (const int*)d_in[16];

  _Float16* ws = (_Float16*)d_ws;
  const _Float16* vw0t = ws + WS_VW0T;
  const _Float16* vw1t = ws + WS_VW1T;
  const _Float16* cw0t = ws + WS_CW0T;
  const _Float16* cw1t = ws + WS_CW1T;

  float* logits = (float*)d_out;
  float* ctx    = logits + (size_t)B_ * L_ * C_;

  // 1) weight images (f16, transposed)
  prep_weights<<<(WS_TOTAL + 255) / 256, 256, 0, stream>>>(vw0, vw1, cw0, cw1, ws);

  // 2) sequential CDE integration: 8 blocks (one per 16-batch-row tile)
  cde_integrate<<<B_ / 16, NT_, 0, stream>>>(ts, ts_interp, obs_interp,
                                             iw0, ib0, iw1, ib1, vb0, vb1,
                                             vw0t, vw1t, t_max, ctx);

  // 3) batch pooling into ctx[:,:,H:2H]
  pool_kernel<<<L_, H_, 0, stream>>>(valid_mask, ctx);

  // 4) classifier GEMMs -> logits
  classifier_kernel<<<(B_ * L_ / 16) / 4, 128, 0, stream>>>(ctx, cw0t, cw1t, cb0, cb1, logits);
}
